// GNNImpute_79087527788859
// MI455X (gfx1250) — matrix operand order repaired and checked by reading, not compiled, not run on hardware
//
#include <hip/hip_runtime.h>
#include <hip/hip_bf16.h>

// ---------------------------------------------------------------------------
// GNNImpute forward on MI455X (gfx1250, wave32).
// All dense GEMMs -> v_wmma_f32_16x16x32_bf16 (f32 accumulate).
// Double-buffered LDS staging (one barrier per K-step), uniform-guard fast
// path, float2 global loads -> packed bf16 pairs, global_prefetch_b8 lookahead.
// ---------------------------------------------------------------------------

#define N_NODES 20000
#define E_EDGES 160000
#define D_IN    3000
#define H_DIM   512
#define Z_DIM   50
#define N_HEADS 3
#define BN_EPS  1e-5f

typedef __bf16        bf16x16 __attribute__((ext_vector_type(16)));
typedef float         f32x8   __attribute__((ext_vector_type(8)));
typedef unsigned int  u32x8   __attribute__((ext_vector_type(8)));

union FragAB { u32x8 u; bf16x16 v; };

__device__ __forceinline__ unsigned short f2bf(float f) {
  unsigned int u = __float_as_uint(f);
  u += 0x7FFFu + ((u >> 16) & 1u);          // round-to-nearest-even
  return (unsigned short)(u >> 16);
}
__device__ __forceinline__ unsigned int pack_bf2(float a, float b) {
  return (unsigned int)f2bf(a) | ((unsigned int)f2bf(b) << 16);
}

// ---------------------------------------------------------------------------
// GEMM: C[M,N] = A[M,K] * B + bias (optional) with optional ReLU.
// A: f32 row-major (lda).  B: f32; transB==0 -> B[k*ldb+n], transB==1 -> B[n*ldb+k].
// Block tile 128x128, BK=32, 256 threads (8 wave32), wave tile 32x64 (2x4 WMMA).
// Requires: M,N,K even; (lda|ldb)*4 % 8 == 0 (true for all call sites).
// ---------------------------------------------------------------------------
#define BM 128
#define BN 128
#define BK 32

__device__ __forceinline__ void stage_A(const float* __restrict__ A,
                                        unsigned short* asp,
                                        int blockM, int k0, int M, int K,
                                        int lda, int tid)
{
  if (blockM + BM <= M && k0 + BK <= K) {       // uniform fast path
#pragma unroll
    for (int it = 0; it < 8; ++it) {
      int e = tid + it * 256;
      int m = e >> 4, k2 = (e & 15) << 1;
      float2 v = *(const float2*)(A + (long long)(blockM + m) * lda + (k0 + k2));
      *(unsigned int*)&asp[m * BK + k2] = pack_bf2(v.x, v.y);
    }
  } else {                                       // edge tile: guarded
    for (int e = tid; e < BM * (BK / 2); e += 256) {
      int m = e >> 4, k2 = (e & 15) << 1;
      int gm = blockM + m, gk = k0 + k2;
      float2 v; v.x = 0.f; v.y = 0.f;
      if (gm < M && gk < K) v = *(const float2*)(A + (long long)gm * lda + gk);
      *(unsigned int*)&asp[m * BK + k2] = pack_bf2(v.x, v.y);
    }
  }
}

__device__ __forceinline__ void stage_B(const float* __restrict__ B,
                                        unsigned short* bsp,
                                        int blockN, int k0, int N, int K,
                                        int ldb, int transB, int tid)
{
  if (transB) {                                  // B[n*ldb+k]: float2 along k
    if (blockN + BN <= N && k0 + BK <= K) {
#pragma unroll
      for (int it = 0; it < 8; ++it) {
        int e = tid + it * 256;
        int n = e >> 4, k2 = (e & 15) << 1;
        float2 v = *(const float2*)(B + (long long)(blockN + n) * ldb + (k0 + k2));
        *(unsigned int*)&bsp[n * BK + k2] = pack_bf2(v.x, v.y);
      }
    } else {
      for (int e = tid; e < BN * (BK / 2); e += 256) {
        int n = e >> 4, k2 = (e & 15) << 1;
        int gn = blockN + n, gk = k0 + k2;
        float2 v; v.x = 0.f; v.y = 0.f;
        if (gn < N && gk < K) v = *(const float2*)(B + (long long)gn * ldb + gk);
        *(unsigned int*)&bsp[n * BK + k2] = pack_bf2(v.x, v.y);
      }
    }
  } else {                                       // B[k*ldb+n]: float2 along n
    if (blockN + BN <= N && k0 + BK <= K) {
#pragma unroll
      for (int it = 0; it < 8; ++it) {
        int e = tid + it * 256;
        int kk = e >> 6, n2 = (e & 63) << 1;
        float2 v = *(const float2*)(B + (long long)(k0 + kk) * ldb + (blockN + n2));
        bsp[n2 * BK + kk]       = f2bf(v.x);
        bsp[(n2 + 1) * BK + kk] = f2bf(v.y);
      }
    } else {
      for (int e = tid; e < (BN / 2) * BK; e += 256) {
        int kk = e >> 6, n2 = (e & 63) << 1;
        int gk = k0 + kk, gn = blockN + n2;
        float2 v; v.x = 0.f; v.y = 0.f;
        if (gk < K && gn < N) v = *(const float2*)(B + (long long)gk * ldb + gn);
        bsp[n2 * BK + kk]       = f2bf(v.x);
        bsp[(n2 + 1) * BK + kk] = f2bf(v.y);
      }
    }
  }
}

__global__ __launch_bounds__(256)
void k_gemm_bf16_wmma(const float* __restrict__ A, const float* __restrict__ B,
                      float* __restrict__ C, const float* __restrict__ bias,
                      int M, int N, int K, int lda, int ldb, int ldc,
                      int transB, int doRelu)
{
  __shared__ unsigned short As[2][BM * BK];   // [m][k] bf16 bits, double-buffered
  __shared__ unsigned short Bs[2][BN * BK];   // [n][k] bf16 bits (K-contiguous)

  const int tid    = threadIdx.x;
  const int lane   = tid & 31;
  const int wave   = tid >> 5;
  const int mlane  = lane & 15;
  const int half   = lane >> 4;
  const int waveM  = (wave >> 1) * 32;        // 4 wave-rows
  const int waveN  = (wave & 1) * 64;         // 2 wave-cols
  const int blockM = blockIdx.x * BM;
  const int blockN = blockIdx.y * BN;

  const f32x8 vzero = {0.f, 0.f, 0.f, 0.f, 0.f, 0.f, 0.f, 0.f};
  f32x8 acc[2][4];
  for (int i = 0; i < 2; ++i)
    for (int j = 0; j < 4; ++j) acc[i][j] = vzero;

  const int nk = (K + BK - 1) / BK;

  // prologue: stage tile 0 into buffer 0
  stage_A(A, As[0], blockM, 0, M, K, lda, tid);
  stage_B(B, Bs[0], blockN, 0, N, K, ldb, transB, tid);
  __syncthreads();

  for (int t = 0; t < nk; ++t) {
    const int cur = t & 1;

    // ---- stage tile t+1 into the other buffer (overlaps with compute) ----
    if (t + 1 < nk) {
      stage_A(A, As[cur ^ 1], blockM, (t + 1) * BK, M, K, lda, tid);
      stage_B(B, Bs[cur ^ 1], blockN, (t + 1) * BK, N, K, ldb, transB, tid);
    }
    // ---- prefetch tile t+2 (global_prefetch_b8) ----
    if ((t + 2) * BK < K) {
      int gm = blockM + (tid >> 1);
      if (gm < M)
        __builtin_prefetch((const void*)(A + (long long)gm * lda + (t + 2) * BK + ((tid & 1) << 4)), 0, 0);
    }

    // ---- fragments from current buffer, contiguous k order (k = 16*half + element) ----
    const unsigned short* asp = As[cur];
    const unsigned short* bsp = Bs[cur];
    FragAB afrag[2];
    for (int i = 0; i < 2; ++i) {
      const unsigned short* ap = &asp[(waveM + i * 16 + mlane) * BK];
#pragma unroll
      for (int r = 0; r < 8; ++r) {
        // A 16x32: VGPR r holds K pair; base = half*16 + 2*r
        int kb = (half << 4) + (r << 1);
        afrag[i].u[r] = *(const unsigned int*)(ap + kb);
      }
    }
    FragAB bfrag[4];
    for (int j = 0; j < 4; ++j) {
      const unsigned short* bp = &bsp[(waveN + j * 16 + mlane) * BK] + (half << 4);
#pragma unroll
      for (int r = 0; r < 8; ++r)      // B 32x16: lane=N, half selects K 0..15 / 16..31
        bfrag[j].u[r] = *(const unsigned int*)(bp + (r << 1));
    }

#pragma unroll
    for (int i = 0; i < 2; ++i)
#pragma unroll
      for (int j = 0; j < 4; ++j)
        acc[i][j] = __builtin_amdgcn_wmma_f32_16x16x32_bf16(
            false, afrag[i].v, false, bfrag[j].v,
            (short)0, acc[i][j], false, false);

    __syncthreads();   // staging of t+1 done everywhere; reads of buf cur done
  }

  // ---- epilogue: bias + optional ReLU, bounds-guarded ----
  for (int i = 0; i < 2; ++i) {
    for (int j = 0; j < 4; ++j) {
      int gn = blockN + waveN + j * 16 + mlane;
      if (gn >= N) continue;
      float bv = bias ? bias[gn] : 0.0f;
#pragma unroll
      for (int vv = 0; vv < 8; ++vv) {   // C/D: row = vgpr + 8*half, col = lane%16
        int gm = blockM + waveM + i * 16 + vv + (half << 3);
        if (gm < M) {
          float r = acc[i][j][vv] + bv;
          if (doRelu) r = fmaxf(r, 0.0f);
          C[(long long)gm * ldc + gn] = r;
        }
      }
    }
  }
}

// ---------------------------------------------------------------------------
// Elementwise / init helpers
// ---------------------------------------------------------------------------
__global__ void k_zero_f32(float* p, long long n) {
  long long i = (long long)blockIdx.x * blockDim.x + threadIdx.x;
  if (i < n) p[i] = 0.0f;
}
__global__ void k_fill_bits(unsigned int* p, unsigned int bits, long long n) {
  long long i = (long long)blockIdx.x * blockDim.x + threadIdx.x;
  if (i < n) p[i] = bits;
}

// ---------------------------------------------------------------------------
// GAT attention scores: one wave32 per (node, head), coalesced lane-strided
// loads + shfl_xor butterfly reduction.
// ---------------------------------------------------------------------------
__global__ __launch_bounds__(256)
void k_att_scores(const float* __restrict__ h,
                  const float* __restrict__ ws,
                  const float* __restrict__ wd,
                  float* __restrict__ a_s, float* __restrict__ a_d,
                  int n, int c)
{
  int wid  = blockIdx.x * (blockDim.x >> 5) + (threadIdx.x >> 5);
  int lane = threadIdx.x & 31;
  if (wid >= n * N_HEADS) return;               // uniform per wave
  int node = wid / N_HEADS, hh = wid % N_HEADS;
  const float* hp = h + (long long)node * (N_HEADS * c) + hh * c;
  const float* s  = ws + hh * c;
  const float* d  = wd + hh * c;
  float ss = 0.f, sd = 0.f;
  for (int i = lane; i < c; i += 32) { float v = hp[i]; ss += v * s[i]; sd += v * d[i]; }
#pragma unroll
  for (int off = 16; off > 0; off >>= 1) {
    ss += __shfl_xor(ss, off, 32);
    sd += __shfl_xor(sd, off, 32);
  }
  if (lane == 0) { a_s[wid] = ss; a_d[wid] = sd; }
}

__device__ __forceinline__ void edge_nodes(const long long* ei, long long E,
                                           long long e, int* s, int* d) {
  if (e < E) { *s = (int)ei[e]; *d = (int)ei[E + e]; }
  else       { int v = (int)(e - E); *s = v; *d = v; }   // self-loops
}

__device__ __forceinline__ float lrelu02(float v) { return v > 0.f ? v : 0.2f * v; }

// segment max via monotonic int/uint atomic trick (init bits = 0xFF800000 = -inf)
__global__ void k_edge_max(const long long* __restrict__ ei, long long E, long long Etot,
                           const float* __restrict__ a_s, const float* __restrict__ a_d,
                           unsigned int* __restrict__ mx)
{
  long long e = (long long)blockIdx.x * blockDim.x + threadIdx.x;
  if (e >= Etot) return;
  int s, d; edge_nodes(ei, E, e, &s, &d);
#pragma unroll
  for (int hh = 0; hh < N_HEADS; ++hh) {
    float v = lrelu02(a_s[s * N_HEADS + hh] + a_d[d * N_HEADS + hh]);
    unsigned int* addr = &mx[(long long)d * N_HEADS + hh];
    if (v >= 0.0f) atomicMax((int*)addr, __float_as_int(v));
    else           atomicMin(addr, __float_as_uint(v));
  }
}

__global__ void k_edge_expsum(const long long* __restrict__ ei, long long E, long long Etot,
                              const float* __restrict__ a_s, const float* __restrict__ a_d,
                              const unsigned int* __restrict__ mx,
                              float* __restrict__ exb, float* __restrict__ den)
{
  long long e = (long long)blockIdx.x * blockDim.x + threadIdx.x;
  if (e >= Etot) return;
  int s, d; edge_nodes(ei, E, e, &s, &d);
#pragma unroll
  for (int hh = 0; hh < N_HEADS; ++hh) {
    float v = lrelu02(a_s[s * N_HEADS + hh] + a_d[d * N_HEADS + hh]);
    float m = __uint_as_float(mx[(long long)d * N_HEADS + hh]);
    float ex = expf(v - m);
    exb[e * N_HEADS + hh] = ex;
    atomicAdd(&den[(long long)d * N_HEADS + hh], ex);
  }
}

// scatter: agg[dst,h,:] += h_src[src,h,:] * alpha   (one block per edge)
__global__ void k_edge_agg(const long long* __restrict__ ei, long long E, long long Etot,
                           const float* __restrict__ hsrc, const float* __restrict__ exb,
                           const float* __restrict__ den, float* __restrict__ agg, int c)
{
  long long e = blockIdx.x;
  if (e >= Etot) return;
  int s, d; edge_nodes(ei, E, e, &s, &d);
  for (int hh = 0; hh < N_HEADS; ++hh) {
    float alpha = exb[e * N_HEADS + hh] / den[(long long)d * N_HEADS + hh];
    const float* hp = hsrc + ((long long)s * N_HEADS + hh) * c;
    float*       op = agg  + ((long long)d * N_HEADS + hh) * c;
    for (int i = threadIdx.x; i < c; i += blockDim.x)
      atomicAdd(&op[i], hp[i] * alpha);
  }
}

// mean over heads + bias -> g[n,c]
__global__ void k_head_mean_bias(const float* __restrict__ agg,
                                 const float* __restrict__ bias,
                                 float* __restrict__ g, int n, int c)
{
  long long t = (long long)blockIdx.x * blockDim.x + threadIdx.x;
  if (t >= (long long)n * c) return;
  int node = (int)(t / c), cc = (int)(t % c);
  float s = 0.f;
#pragma unroll
  for (int hh = 0; hh < N_HEADS; ++hh)
    s += agg[((long long)node * N_HEADS + hh) * c + cc];
  g[t] = s * (1.0f / N_HEADS) + bias[cc];
}

// BatchNorm column stats (training-mode, biased var) : one block per column
__global__ __launch_bounds__(256)
void k_bn_stats(const float* __restrict__ g, float* __restrict__ mean,
                float* __restrict__ rstd, int n, int C)
{
  __shared__ float s1[256], s2[256];
  int c = blockIdx.x;
  float a = 0.f, b = 0.f;
  for (int r = threadIdx.x; r < n; r += 256) {
    float v = g[(long long)r * C + c];
    a += v; b += v * v;
  }
  s1[threadIdx.x] = a; s2[threadIdx.x] = b;
  __syncthreads();
  for (int st = 128; st > 0; st >>= 1) {
    if (threadIdx.x < st) { s1[threadIdx.x] += s1[threadIdx.x + st];
                            s2[threadIdx.x] += s2[threadIdx.x + st]; }
    __syncthreads();
  }
  if (threadIdx.x == 0) {
    float m = s1[0] / n;
    float var = s2[0] / n - m * m;
    mean[c] = m;
    rstd[c] = rsqrtf(var + BN_EPS);
  }
}

__global__ void k_bn_apply_relu(float* __restrict__ g,
                                const float* __restrict__ mean,
                                const float* __restrict__ rstd,
                                const float* __restrict__ gamma,
                                const float* __restrict__ beta,
                                int n, int C)
{
  long long t = (long long)blockIdx.x * blockDim.x + threadIdx.x;
  if (t >= (long long)n * C) return;
  int cc = (int)(t % C);
  float v = (g[t] - mean[cc]) * rstd[cc] * gamma[cc] + beta[cc];
  g[t] = fmaxf(v, 0.0f);
}

// ---------------------------------------------------------------------------
// Orchestration
// ---------------------------------------------------------------------------
static inline int cdiv(long long a, long long b) { return (int)((a + b - 1) / b); }

extern "C" void kernel_launch(void* const* d_in, const int* in_sizes, int n_in,
                              void* d_out, int out_size, void* d_ws, size_t ws_size,
                              hipStream_t stream)
{
  (void)in_sizes; (void)n_in; (void)out_size; (void)ws_size;

  const float*     x        = (const float*)d_in[0];
  const long long* ei       = (const long long*)d_in[1];   // int64 [2,E]
  const float*     W1       = (const float*)d_in[2];
  const float*     att_src1 = (const float*)d_in[3];
  const float*     att_dst1 = (const float*)d_in[4];
  const float*     b1       = (const float*)d_in[5];
  const float*     bn1_g    = (const float*)d_in[6];
  const float*     bn1_b    = (const float*)d_in[7];
  const float*     W2       = (const float*)d_in[8];
  const float*     att_src2 = (const float*)d_in[9];
  const float*     att_dst2 = (const float*)d_in[10];
  const float*     b2       = (const float*)d_in[11];
  const float*     bn2_g    = (const float*)d_in[12];
  const float*     bn2_b    = (const float*)d_in[13];
  const float*     lin1_w   = (const float*)d_in[14];      // [H,Z]
  const float*     lin1_b   = (const float*)d_in[15];
  const float*     dbn_g    = (const float*)d_in[16];
  const float*     dbn_b    = (const float*)d_in[17];
  const float*     lin2_w   = (const float*)d_in[18];      // [D,H]
  const float*     lin2_b   = (const float*)d_in[19];
  float*           out      = (float*)d_out;

  const int       Nn = N_NODES, HC1 = N_HEADS * H_DIM, HC2 = N_HEADS * Z_DIM;
  const long long Ee = E_EDGES, Etot = (long long)E_EDGES + N_NODES;

  // ---- workspace layout (bytes) ----
  char* w = (char*)d_ws;
  float* buf0 = (float*)(w);                         // 122,880,000 B : h1 / h2 / (z at +32MB)
  float* buf1 = (float*)(w + 122880000LL);           // 122,880,000 B : agg1 / agg2
  float* buf2 = (float*)(w + 245760000LL);           //  40,960,000 B : g1->h / t3->d
  char*  sm   =          w + 286720000LL;            // small region
  float*        a_s  = (float*)(sm);                 // 240 KB
  float*        a_d  = (float*)(sm +   240000);
  unsigned int* mx   = (unsigned int*)(sm + 480000);
  float*        den  = (float*)(sm +   720000);
  float*        exb  = (float*)(sm +   960000);      // 2.16 MB
  float*        mean = (float*)(sm +  3200000);
  float*        rstd = (float*)(sm +  3300000);
  float*        zb   = (float*)(w + 33554432LL);     // z [N,Z] inside buf0 region (h2 ends at 12MB)

  const long long nh = (long long)Nn * N_HEADS;      // 60000

  // ================= Layer 1: GAT(x; W1) =================
  {
    dim3 g(cdiv(Nn, BM), cdiv(HC1, BN));
    k_gemm_bf16_wmma<<<g, 256, 0, stream>>>(x, W1, buf0, nullptr,
        Nn, HC1, D_IN, D_IN, HC1, HC1, 0, 0);                     // h1 = x@W1
  }
  k_att_scores<<<cdiv(nh * 32, 256), 256, 0, stream>>>(buf0, att_src1, att_dst1,
                                                       a_s, a_d, Nn, H_DIM);
  k_fill_bits<<<cdiv(nh, 256), 256, 0, stream>>>(mx, 0xFF800000u, nh);
  k_zero_f32<<<cdiv(nh, 256), 256, 0, stream>>>(den, nh);
  {
    long long na = (long long)Nn * N_HEADS * H_DIM;               // 30.72M
    k_zero_f32<<<cdiv(na, 256), 256, 0, stream>>>(buf1, na);
  }
  k_edge_max   <<<cdiv(Etot, 256), 256, 0, stream>>>(ei, Ee, Etot, a_s, a_d, mx);
  k_edge_expsum<<<cdiv(Etot, 256), 256, 0, stream>>>(ei, Ee, Etot, a_s, a_d, mx, exb, den);
  k_edge_agg   <<<(int)Etot, 256, 0, stream>>>(ei, Ee, Etot, buf0, exb, den, buf1, H_DIM);
  {
    long long ng = (long long)Nn * H_DIM;
    k_head_mean_bias<<<cdiv(ng, 256), 256, 0, stream>>>(buf1, b1, buf2, Nn, H_DIM);
    k_bn_stats<<<H_DIM, 256, 0, stream>>>(buf2, mean, rstd, Nn, H_DIM);
    k_bn_apply_relu<<<cdiv(ng, 256), 256, 0, stream>>>(buf2, mean, rstd, bn1_g, bn1_b, Nn, H_DIM);
  }                                                               // buf2 = h [N,512]

  // ================= Layer 2: GAT(h; W2) =================
  {
    dim3 g(cdiv(Nn, BM), cdiv(HC2, BN));
    k_gemm_bf16_wmma<<<g, 256, 0, stream>>>(buf2, W2, buf0, nullptr,
        Nn, HC2, H_DIM, H_DIM, HC2, HC2, 0, 0);                   // h2 = h@W2
  }
  k_att_scores<<<cdiv(nh * 32, 256), 256, 0, stream>>>(buf0, att_src2, att_dst2,
                                                       a_s, a_d, Nn, Z_DIM);
  k_fill_bits<<<cdiv(nh, 256), 256, 0, stream>>>(mx, 0xFF800000u, nh);
  k_zero_f32<<<cdiv(nh, 256), 256, 0, stream>>>(den, nh);
  {
    long long na = (long long)Nn * N_HEADS * Z_DIM;               // 3M
    k_zero_f32<<<cdiv(na, 256), 256, 0, stream>>>(buf1, na);
  }
  k_edge_max   <<<cdiv(Etot, 256), 256, 0, stream>>>(ei, Ee, Etot, a_s, a_d, mx);
  k_edge_expsum<<<cdiv(Etot, 256), 256, 0, stream>>>(ei, Ee, Etot, a_s, a_d, mx, exb, den);
  k_edge_agg   <<<(int)Etot, 64, 0, stream>>>(ei, Ee, Etot, buf0, exb, den, buf1, Z_DIM);
  {
    long long ng = (long long)Nn * Z_DIM;
    k_head_mean_bias<<<cdiv(ng, 256), 256, 0, stream>>>(buf1, b2, zb, Nn, Z_DIM);
    k_bn_stats<<<Z_DIM, 256, 0, stream>>>(zb, mean, rstd, Nn, Z_DIM);
    k_bn_apply_relu<<<cdiv(ng, 256), 256, 0, stream>>>(zb, mean, rstd, bn2_g, bn2_b, Nn, Z_DIM);
  }                                                               // zb = z [N,50]

  // ================= Decoder =================
  {
    dim3 g(cdiv(Nn, BM), cdiv(H_DIM, BN));
    k_gemm_bf16_wmma<<<g, 256, 0, stream>>>(zb, lin1_w, buf2, lin1_b,
        Nn, H_DIM, Z_DIM, Z_DIM, Z_DIM, H_DIM, 1, 0);             // t3 = z@lin1^T + b
    long long ng = (long long)Nn * H_DIM;
    k_bn_stats<<<H_DIM, 256, 0, stream>>>(buf2, mean, rstd, Nn, H_DIM);
    k_bn_apply_relu<<<cdiv(ng, 256), 256, 0, stream>>>(buf2, mean, rstd, dbn_g, dbn_b, Nn, H_DIM);
  }                                                               // buf2 = d [N,512]
  {
    dim3 g(cdiv(Nn, BM), cdiv(D_IN, BN));
    k_gemm_bf16_wmma<<<g, 256, 0, stream>>>(buf2, lin2_w, out, lin2_b,
        Nn, D_IN, H_DIM, H_DIM, H_DIM, D_IN, 1, 1);               // out = relu(d@lin2^T + b)
  }
}
